// SmallContConvWithMLPKernel_28269474742570
// MI455X (gfx1250) — matrix operand order repaired and checked by reading, not compiled, not run on hardware
//
#include <hip/hip_runtime.h>
#include <cstdint>

#define NPTS 100000
#define KNBR 32
#define WID  32

// Persistent-wave decomposition: 1250 blocks * 8 waves = 10000 waves,
// each wave handles exactly NPTS/10000 = 10 nodes (uniform -> no divergence).
#define BLOCKS 1250
#define WPB    8
#define TOTALW (BLOCKS * WPB)
#define ITERS  (NPTS / TOTALW)

typedef __attribute__((ext_vector_type(16))) __bf16 v16bf;
typedef __attribute__((ext_vector_type(2)))  __bf16 v2bf;
typedef __attribute__((ext_vector_type(8)))  float  v8f;

// Wave-local LDS sync: per-wave private LDS slices only need this wave's DS ops
// to complete (DS is in-order per wave; DScnt covers all 32 lanes of each op).
__device__ __forceinline__ void wave_lds_sync() {
  asm volatile("s_wait_dscnt 0x0" ::: "memory");
}

// Pack two floats into one dword of bf16 pairs (lowers to v_cvt_pk_bf16_f32).
__device__ __forceinline__ uint32_t pkbf(float a, float b) {
  v2bf p;
  p[0] = (__bf16)a;
  p[1] = (__bf16)b;
  uint32_t u;
  __builtin_memcpy(&u, &p, 4);
  return u;
}

// ---- fast GELU (tanh form): ~1e-3 abs accurate, uses v_exp_f32 + v_rcp_f32 ----
__device__ __forceinline__ float fast_gelu(float x) {
  float u = 0.7978845608028654f * (x + 0.044715f * x * x * x);
  float e = __expf(2.0f * u);
  float t = 1.0f - 2.0f * __builtin_amdgcn_rcpf(e + 1.0f);
  return 0.5f * x * (1.0f + t);
}

__device__ __forceinline__ v8f wmma_bf16(v16bf a, v16bf b, v8f c) {
  // D = A(16x32) * B(32x16) + C, f32 accumulate
  return __builtin_amdgcn_wmma_f32_16x16x32_bf16(false, a, false, b, (short)0, c,
                                                 false, false);
}

// B-operand (32x16 K-major tile) striping: lane l holds column n = nbase+(l&15),
// khalf = 8*(l>>4): elements j<8 -> k = kbase + khalf + j, j>=8 -> k = kbase + 16 + khalf + (j-8). Zero-pad k>=kvalid.
// Loaded ONCE per wave, amortized over ITERS nodes.
__device__ __forceinline__ v16bf load_btile(const float* __restrict__ w, int kbase,
                                            int nbase, int kvalid, int lane) {
  const int khalf = (lane >> 4) << 3;
  const int n = nbase + (lane & 15);
  uint32_t u[8];
#pragma unroll
  for (int j = 0; j < 16; j += 2) {
    const int k0 = kbase + khalf + ((j < 8) ? j : (j + 8));
    const int k1 = k0 + 1;
    const float f0 = (k0 < kvalid) ? w[k0 * WID + n] : 0.0f;
    const float f1 = (k1 < kvalid) ? w[k1 * WID + n] : 0.0f;
    u[j >> 1] = pkbf(f0, f1);
  }
  v16bf out;
  __builtin_memcpy(&out, u, sizeof(out));
  return out;
}

// A-operand (16x32 M-major tile) from an LDS row-major bf16 buffer.
// Lane l: row m, low 8 elems at k = 8*(l>>4), high 8 at k = 16 + 8*(l>>4):
// two contiguous 16B chunks -> two ds_load_b128.
__device__ __forceinline__ v16bf load_atile(const __bf16* base, int strideElems,
                                            int mrow, int kElemOff, int lane) {
  const __bf16* p = base + mrow * strideElems + kElemOff + ((lane >> 4) << 3);
  uint4 lo = *(const uint4*)(p);
  uint4 hi = *(const uint4*)(p + 16);
  uint32_t u[8] = {lo.x, lo.y, lo.z, lo.w, hi.x, hi.y, hi.z, hi.w};
  v16bf out;
  __builtin_memcpy(&out, u, sizeof(out));
  return out;
}

// ============================ conv1: [N,3] -> [N,32] bf16 ============================
__global__ __launch_bounds__(256) void conv1_kernel(
    const float* __restrict__ x0, const float* __restrict__ W1a,
    const float* __restrict__ b1a, const float* __restrict__ W2a,
    const float* __restrict__ b2a, const int* __restrict__ nbr,
    __bf16* __restrict__ x1out) {
  constexpr int S1 = 40, S2 = 40;  // padded row strides (80B: 16B aligned, bank-spread)
  __shared__ __align__(16) __bf16 lds1[WPB][32 * S1];
  __shared__ __align__(16) __bf16 lds2[WPB][32 * S2];
  const int lane = threadIdx.x & 31;
  const int wv = threadIdx.x >> 5;
  const int wid = blockIdx.x * WPB + wv;

  // Per-wave setup (amortized over ITERS nodes):
  v16bf B1[2], B2[2];
  B1[0] = load_btile(W1a, 0, 0, 6, lane);
  B1[1] = load_btile(W1a, 0, 16, 6, lane);
  B2[0] = load_btile(W2a, 0, 0, 32, lane);
  B2[1] = load_btile(W2a, 0, 16, 32, lane);
  const float seed1[2] = {b1a[lane & 15], b1a[16 + (lane & 15)]};
  const float seed2[2] = {b2a[lane & 15], b2a[16 + (lane & 15)]};

  const uint4 zq = {0u, 0u, 0u, 0u};
  for (int it = 0; it < ITERS; ++it) {
    const int i = wid + it * TOTALW;
    // Always-valid clamped prefetch of next node's index line (no loop peeling).
    const int ipf = (i + TOTALW < NPTS) ? (i + TOTALW) : (NPTS - 1);
    __builtin_prefetch(&nbr[ipf * KNBR], 0, 1);

    // Stage edge features: row e=lane = [nb(3) | self(3) | zeros...] (zero-padded K).
    const int nb = nbr[i * KNBR + lane];
    const float n0 = x0[nb * 3 + 0], n1 = x0[nb * 3 + 1], n2 = x0[nb * 3 + 2];
    const float s0f = x0[i * 3 + 0], s1f = x0[i * 3 + 1], s2f = x0[i * 3 + 2];
    uint4 first;
    first.x = pkbf(n0, n1);
    first.y = pkbf(n2, s0f);
    first.z = pkbf(s1f, s2f);
    first.w = 0u;
    __bf16* r = &lds1[wv][lane * S1];
    *(uint4*)(r + 0)  = first;
    *(uint4*)(r + 8)  = zq;
    *(uint4*)(r + 16) = zq;
    *(uint4*)(r + 24) = zq;
    wave_lds_sync();

    // GEMM1: (32 edges x 32K) @ (32K x 32) -> C1, K>=6 contributes zero.
    v8f C1[2][2];
#pragma unroll
    for (int mt = 0; mt < 2; ++mt) {
      v16bf A = load_atile(&lds1[wv][0], S1, mt * 16 + (lane & 15), 0, lane);
#pragma unroll
      for (int nt = 0; nt < 2; ++nt) {
        v8f c;
#pragma unroll
        for (int q = 0; q < 8; ++q) c[q] = seed1[nt];
        C1[mt][nt] = wmma_bf16(A, B1[nt], c);
      }
    }

    // GELU in C layout, restage as bf16 A matrix for GEMM2.
#pragma unroll
    for (int mt = 0; mt < 2; ++mt)
#pragma unroll
      for (int nt = 0; nt < 2; ++nt)
#pragma unroll
        for (int q = 0; q < 8; ++q) {
          const int m = mt * 16 + ((lane >> 4) << 3) + q;
          const int n = nt * 16 + (lane & 15);
          lds2[wv][m * S2 + n] = (__bf16)fast_gelu(C1[mt][nt][q]);
        }
    wave_lds_sync();

    // GEMM2: (32x32) @ (32x32) -> C2.
    v8f C2[2][2];
#pragma unroll
    for (int mt = 0; mt < 2; ++mt) {
      v16bf A = load_atile(&lds2[wv][0], S2, mt * 16 + (lane & 15), 0, lane);
#pragma unroll
      for (int nt = 0; nt < 2; ++nt) {
        v8f c;
#pragma unroll
        for (int q = 0; q < 8; ++q) c[q] = seed2[nt];
        C2[mt][nt] = wmma_bf16(A, B2[nt], c);
      }
    }

    // Mean over 32 edge rows -> x1[i, :] (bf16 for conv2 gathers).
#pragma unroll
    for (int nt = 0; nt < 2; ++nt) {
      float s = 0.0f;
#pragma unroll
      for (int q = 0; q < 8; ++q) s += C2[0][nt][q] + C2[1][nt][q];
      s += __shfl_xor(s, 16, 32);
      s *= (1.0f / 32.0f);
      if ((lane >> 4) == nt) x1out[i * WID + lane] = (__bf16)s;
    }
  }
}

// ================= conv2 + output: [N,32] bf16 -> [N,1] f32 (fused) =================
__global__ __launch_bounds__(256) void conv2_kernel(
    const __bf16* __restrict__ x1, const float* __restrict__ W1b,
    const float* __restrict__ b1b, const float* __restrict__ W2b,
    const float* __restrict__ b2b, const float* __restrict__ Wo,
    const float* __restrict__ bo, const int* __restrict__ nbr,
    float* __restrict__ out) {
  constexpr int S1 = 72, S2 = 40;  // 144B / 80B padded strides
  __shared__ __align__(16) __bf16 lds1[WPB][32 * S1];
  __shared__ __align__(16) __bf16 lds2[WPB][32 * S2];
  const int lane = threadIdx.x & 31;
  const int wv = threadIdx.x >> 5;
  const int wid = blockIdx.x * WPB + wv;

  // Per-wave setup (amortized over ITERS nodes):
  v16bf B1[2][2], B2[2];
#pragma unroll
  for (int kc = 0; kc < 2; ++kc)
#pragma unroll
    for (int nt = 0; nt < 2; ++nt)
      B1[kc][nt] = load_btile(W1b, kc * 32, nt * 16, 64, lane);
  B2[0] = load_btile(W2b, 0, 0, 32, lane);
  B2[1] = load_btile(W2b, 0, 16, 32, lane);
  const float seed1[2] = {b1b[lane & 15], b1b[16 + (lane & 15)]};
  const float seed2[2] = {b2b[lane & 15], b2b[16 + (lane & 15)]};
  const float wo_l = Wo[lane];
  const float bo0 = bo[0];

  for (int it = 0; it < ITERS; ++it) {
    const int i = wid + it * TOTALW;
    const int ipf = (i + TOTALW < NPTS) ? (i + TOTALW) : (NPTS - 1);
    __builtin_prefetch(&nbr[ipf * KNBR], 0, 1);

    // Gather: edge e=lane pulls 64B neighbor row + 64B self row (L2-resident).
    const int nb = nbr[i * KNBR + lane];
    const uint4* nrow = (const uint4*)(x1 + nb * WID);
    const uint4* srow = (const uint4*)(x1 + i * WID);
    __bf16* r = &lds1[wv][lane * S1];
    *(uint4*)(r + 0)  = nrow[0];
    *(uint4*)(r + 8)  = nrow[1];
    *(uint4*)(r + 16) = nrow[2];
    *(uint4*)(r + 24) = nrow[3];
    *(uint4*)(r + 32) = srow[0];
    *(uint4*)(r + 40) = srow[1];
    *(uint4*)(r + 48) = srow[2];
    *(uint4*)(r + 56) = srow[3];
    wave_lds_sync();

    // GEMM1: (32 edges x 64K) @ (64K x 32): two K-chunks chained into accumulator.
    v8f C1[2][2];
#pragma unroll
    for (int mt = 0; mt < 2; ++mt) {
      v16bf A0 = load_atile(&lds1[wv][0], S1, mt * 16 + (lane & 15), 0, lane);
      v16bf A1 = load_atile(&lds1[wv][0], S1, mt * 16 + (lane & 15), 32, lane);
#pragma unroll
      for (int nt = 0; nt < 2; ++nt) {
        v8f c;
#pragma unroll
        for (int q = 0; q < 8; ++q) c[q] = seed1[nt];
        c = wmma_bf16(A0, B1[0][nt], c);
        C1[mt][nt] = wmma_bf16(A1, B1[1][nt], c);
      }
    }

    // GELU + restage for GEMM2.
#pragma unroll
    for (int mt = 0; mt < 2; ++mt)
#pragma unroll
      for (int nt = 0; nt < 2; ++nt)
#pragma unroll
        for (int q = 0; q < 8; ++q) {
          const int m = mt * 16 + ((lane >> 4) << 3) + q;
          const int n = nt * 16 + (lane & 15);
          lds2[wv][m * S2 + n] = (__bf16)fast_gelu(C1[mt][nt][q]);
        }
    wave_lds_sync();

    v8f C2[2][2];
#pragma unroll
    for (int mt = 0; mt < 2; ++mt) {
      v16bf A = load_atile(&lds2[wv][0], S2, mt * 16 + (lane & 15), 0, lane);
#pragma unroll
      for (int nt = 0; nt < 2; ++nt) {
        v8f c;
#pragma unroll
        for (int q = 0; q < 8; ++q) c[q] = seed2[nt];
        C2[mt][nt] = wmma_bf16(A, B2[nt], c);
      }
    }

    // Column means over 32 edges, then fused gelu(x2) @ Wo + bo via wave reduction.
    float s0 = 0.0f, s1 = 0.0f;
#pragma unroll
    for (int q = 0; q < 8; ++q) {
      s0 += C2[0][0][q] + C2[1][0][q];
      s1 += C2[0][1][q] + C2[1][1][q];
    }
    s0 += __shfl_xor(s0, 16, 32);
    s1 += __shfl_xor(s1, 16, 32);
    // lane < 16 owns channel c = lane (N-tile 0); lane >= 16 owns c = lane (N-tile 1).
    const float xv = ((lane < 16) ? s0 : s1) * (1.0f / 32.0f);
    float contrib = fast_gelu(xv) * wo_l;
#pragma unroll
    for (int off = 16; off > 0; off >>= 1) contrib += __shfl_xor(contrib, off, 32);
    if (lane == 0) out[i] = contrib + bo0;
  }
}

extern "C" void kernel_launch(void* const* d_in, const int* in_sizes, int n_in,
                              void* d_out, int out_size, void* d_ws, size_t ws_size,
                              hipStream_t stream) {
  (void)in_sizes; (void)n_in; (void)out_size; (void)ws_size;
  const float* x_in = (const float*)d_in[0];
  const float* W1a  = (const float*)d_in[1];
  const float* b1a  = (const float*)d_in[2];
  const float* W2a  = (const float*)d_in[3];
  const float* b2a  = (const float*)d_in[4];
  const float* W1b  = (const float*)d_in[5];
  const float* b1b  = (const float*)d_in[6];
  const float* W2b  = (const float*)d_in[7];
  const float* b2b  = (const float*)d_in[8];
  const float* Wo   = (const float*)d_in[9];
  const float* bo   = (const float*)d_in[10];
  const int*   nbr  = (const int*)d_in[11];

  __bf16* x1 = (__bf16*)d_ws;          // N*32 bf16 = 6.4 MB scratch
  float*  out = (float*)d_out;

  dim3 blk(256);
  dim3 grd(BLOCKS);  // persistent waves: 10000 waves x 10 nodes each
  conv1_kernel<<<grd, blk, 0, stream>>>(x_in, W1a, b1a, W2a, b2a, nbr, x1);
  conv2_kernel<<<grd, blk, 0, stream>>>(x1, W1b, b1b, W2b, b2b, Wo, bo, nbr, out);
}
